// FGA_C_22479858828040
// MI455X (gfx1250) — compile-verified, matrix operand order checked by reading, not run on hardware
//
#include <hip/hip_runtime.h>
#include <hip/hip_bf16.h>
#include <math.h>

typedef __bf16 bf16_t;
typedef __attribute__((ext_vector_type(16))) __bf16 v16bf;
typedef __attribute__((ext_vector_type(8)))  __bf16 v8bf;
typedef __attribute__((ext_vector_type(8)))  float  v8f;

#define HEADS 8

// ---------- WMMA helpers (CDNA5 16x16x32 bf16, fp32 accum) ----------
__device__ __forceinline__ v8f v8f_zero() {
  v8f z;
#pragma unroll
  for (int i = 0; i < 8; ++i) z[i] = 0.0f;
  return z;
}

__device__ __forceinline__ v8f wmma_bf16(v16bf a, v16bf b, v8f c) {
  return __builtin_amdgcn_wmma_f32_16x16x32_bf16(false, a, false, b, (short)0, c,
                                                 false, false);
}

// Fragment where this lane's row/column is contiguous along K (bf16).
// lane<16: K = {0..7, 16..23}; lane>=16: K = {8..15, 24..31} (ISA 16-bit layout)
// Two 16-byte loads (global_load_b128 / ds_load_b128).
__device__ __forceinline__ v16bf frag_contig(const bf16_t* row, int k0, int hi) {
  v8bf lo  = *(const v8bf*)(row + k0 + hi * 8);
  v8bf hi8 = *(const v8bf*)(row + k0 + 16 + hi * 8);
  v16bf r;
#pragma unroll
  for (int e = 0; e < 8; ++e) { r[e] = lo[e]; r[8 + e] = hi8[e]; }
  return r;
}

__device__ __forceinline__ v16bf frag_contig_f32(const float* row, int k0, int hi) {
  v16bf r;
#pragma unroll
  for (int e = 0; e < 8; ++e) {
    r[e]     = (bf16_t)row[k0 + hi * 8 + e];
    r[8 + e] = (bf16_t)row[k0 + 16 + hi * 8 + e];
  }
  return r;
}

// ---------- 1. sort init ----------
__global__ void k_init_sort(const float* __restrict__ idxm, float* keys, int* vals,
                            int L, int Lpad) {
  int t = blockIdx.x * blockDim.x + threadIdx.x;
  int b = t / Lpad, i = t % Lpad;
  keys[t] = (i < L) ? idxm[b * L + i] : 1e30f;
  vals[t] = i;
}

// ---------- 2. bitonic pass ----------
__global__ void k_bitonic(float* keys, int* vals, int j, int k, int Lpad) {
  int t = blockIdx.x * blockDim.x + threadIdx.x;
  int b = t / Lpad, i = t % Lpad;
  int ixj = i ^ j;
  if (ixj > i) {
    float* kb = keys + (size_t)b * Lpad;
    int*   vb = vals + (size_t)b * Lpad;
    bool up = ((i & k) == 0);
    float a = kb[i], c = kb[ixj];
    bool sw = up ? (a > c) : (a < c);
    if (sw) {
      kb[i] = c; kb[ixj] = a;
      int tv = vb[i]; vb[i] = vb[ixj]; vb[ixj] = tv;
    }
  }
}

// ---------- 3. fused scale + gather to bf16 ----------
__global__ void k_gather(const float* __restrict__ feats, const float* __restrict__ idxm,
                         const int* __restrict__ vals, bf16_t* xb, int L, int Lpad, int C) {
  size_t t = (size_t)blockIdx.x * blockDim.x + threadIdx.x;  // over B*C*L
  int l = (int)(t % L);
  size_t bc = t / L;
  int c = (int)(bc % C);
  int b = (int)(bc / C);
  int p = vals[(size_t)b * Lpad + l];
  float v = feats[((size_t)b * C + c) * L + p] * idxm[(size_t)b * L + p];
  xb[t] = (bf16_t)v;
}

// ---------- 4. qkv 1x1 conv as WMMA GEMM: Y(960,L) = W(960,192) @ X(192,L) + b ----
// B tile staged TRANSPOSED in LDS: Bs[n][k], pitch 40 -> fragment = 2x ds_load_b128
__global__ void k_gemm_qkv(const float* __restrict__ W, const float* __restrict__ bias,
                           const bf16_t* __restrict__ X, bf16_t* __restrict__ Y,
                           int Mtot, int K, int L) {
  __shared__ bf16_t Bs[64 * 40];
  const int PITCH = 40;
  int b  = blockIdx.z;
  int n0 = blockIdx.x * 64;
  int m0 = blockIdx.y * 128;
  int wave = threadIdx.x >> 5, lane = threadIdx.x & 31;
  int hi = lane >> 4, lx = lane & 15;
  int mw = m0 + wave * 16;
  bool active = (mw < Mtot);
  const bf16_t* Xb = X + (size_t)b * K * L;
  v8f acc[4];
#pragma unroll
  for (int i = 0; i < 4; ++i) acc[i] = v8f_zero();

  for (int k0 = 0; k0 < K; k0 += 32) {
    __syncthreads();
#pragma unroll
    for (int e = threadIdx.x; e < 32 * 64; e += 256) {
      int kk = e >> 6, nn = e & 63;  // coalesced global read along nn
      Bs[nn * PITCH + kk] = Xb[(size_t)(k0 + kk) * L + n0 + nn];
    }
    if (k0 + 32 < K) __builtin_prefetch(Xb + (size_t)(k0 + 32) * L + n0, 0, 0);
    __syncthreads();
    if (active) {
      const float* wrow = W + (size_t)(mw + lx) * K;
      v16bf af = frag_contig_f32(wrow, k0, hi);
      v16bf bfms[4];
#pragma unroll
      for (int nt = 0; nt < 4; ++nt)
        bfms[nt] = frag_contig(Bs + (nt * 16 + lx) * PITCH, 0, hi);
#pragma unroll
      for (int nt = 0; nt < 4; ++nt)
        acc[nt] = wmma_bf16(af, bfms[nt], acc[nt]);
    }
  }
  if (active) {
#pragma unroll
    for (int nt = 0; nt < 4; ++nt)
#pragma unroll
      for (int r = 0; r < 8; ++r) {
        int m = mw + hi * 8 + r;
        int n = n0 + nt * 16 + lx;
        Y[((size_t)b * Mtot + m) * L + n] = (bf16_t)(acc[nt][r] + bias[m]);
      }
  }
}

// ---------- 5. depthwise 3x3 ----------
__global__ void k_dwconv(const bf16_t* __restrict__ in, const float* __restrict__ w,
                         const float* __restrict__ bias, bf16_t* __restrict__ out,
                         int Ctot, int H, int Wd) {
  size_t t = (size_t)blockIdx.x * blockDim.x + threadIdx.x;  // B*Ctot*H*W
  int x = (int)(t % Wd);
  size_t r1 = t / Wd;
  int y = (int)(r1 % H);
  size_t r2 = r1 / H;
  int c = (int)(r2 % Ctot);
  int b = (int)(r2 / Ctot);
  const bf16_t* ip = in + ((size_t)b * Ctot + c) * H * Wd;
  const float* wc = w + c * 9;
  float s = bias[c];
#pragma unroll
  for (int dy = -1; dy <= 1; ++dy) {
    int yy = y + dy;
    if (yy < 0 || yy >= H) continue;
#pragma unroll
    for (int dx = -1; dx <= 1; ++dx) {
      int xx = x + dx;
      if (xx < 0 || xx >= Wd) continue;
      s += (float)ip[(size_t)yy * Wd + xx] * wc[(dy + 1) * 3 + (dx + 1)];
    }
  }
  out[t] = (bf16_t)s;
}

// ---------- 5b. grouped-transposed copy of V: Vt[b][h][g][cg] = Vg[cg][g] ----------
__global__ void k_vt(const bf16_t* __restrict__ qkv, bf16_t* __restrict__ Vt,
                     int C, int CO, int L, int hwg) {
  size_t t = (size_t)blockIdx.x * blockDim.x + threadIdx.x;  // B*8*hwg*192
  int cg = (int)(t % 192);
  size_t r1 = t / 192;
  int g = (int)(r1 % hwg);
  size_t r2 = r1 / hwg;
  int h = (int)(r2 % 8);
  int b = (int)(r2 / 8);
  const int ch = C / HEADS;  // 24
  int f = cg / ch, cc = cg % ch;
  Vt[t] = qkv[((size_t)b * CO + 4 * C + h * ch + cc) * L + (size_t)f * hwg + g];
}

// ---------- 6. per-group mean of V: gvm[b][f][c] ----------
__global__ void k_gmean(const bf16_t* __restrict__ qkv, float* __restrict__ gvm,
                        int C, int CO, int L, int hwg) {
  int bid = blockIdx.x;  // B*8*C blocks
  int c = bid % C;
  int f = (bid / C) % 8;
  int b = bid / (C * 8);
  const bf16_t* p = qkv + ((size_t)b * CO + 4 * C + c) * L + (size_t)f * hwg;
  float s = 0.f;
  for (int g = threadIdx.x; g < hwg; g += 128) s += (float)p[g];
  __shared__ float red[128];
  red[threadIdx.x] = s;
  __syncthreads();
  for (int o = 64; o > 0; o >>= 1) {
    if (threadIdx.x < o) red[threadIdx.x] += red[threadIdx.x + o];
    __syncthreads();
  }
  if (threadIdx.x == 0) gvm[(b * 8 + f) * C + c] = red[0] / (float)hwg;
}

// ---------- 7. 8x8 cosine + top-2 (second-best) per column ----------
__global__ void k_top2(const float* __restrict__ gvm, int* __restrict__ sim, int C) {
  int b = blockIdx.x;
  __shared__ float cosm[64];
  __shared__ float nrm[8];
  int t = threadIdx.x;  // 64 threads
  if (t < 8) {
    const float* r = gvm + (b * 8 + t) * C;
    float s = 0.f;
    for (int c = 0; c < C; ++c) s += r[c] * r[c];
    nrm[t] = sqrtf(s);
  }
  __syncthreads();
  int i = t >> 3, j = t & 7;
  const float* ri = gvm + (b * 8 + i) * C;
  const float* rj = gvm + (b * 8 + j) * C;
  float d = 0.f;
  for (int c = 0; c < C; ++c) d += ri[c] * rj[c];
  cosm[i * 8 + j] = d / fmaxf(nrm[i] * nrm[j], 1e-8f);
  __syncthreads();
  if (t < 8) {
    float v1 = -1e30f, v2 = -1e30f;
    int i1 = 0, i2 = 0;
    for (int ii = 0; ii < 8; ++ii) {
      float v = cosm[ii * 8 + t];
      if (v > v1) { v2 = v1; i2 = i1; v1 = v; i1 = ii; }
      else if (v > v2) { v2 = v; i2 = ii; }
    }
    sim[b * 8 + t] = i2;
  }
}

// ---------- 8. fused channel attention per (b, head) ----------
// S = norm(Qg) @ norm(Kg)^T * temp ; P = softmax_1(S) ; out = P @ Vg
// cross!=0: query group f remapped via sim[], output accumulated with += para.
__global__ void k_attn(const bf16_t* __restrict__ qkv, const bf16_t* __restrict__ Vt,
                       const float* __restrict__ temperature,
                       const float* __restrict__ para, const int* __restrict__ sim,
                       float* __restrict__ y, int C, int CO, int L, int hwg,
                       int qoff, int koff, int cross) {
  extern __shared__ char smem[];
  float* S  = (float*)smem;  // 192*192 f32 (rows reused as bf16 P)
  float* qn = S + 192 * 192;
  float* kn = qn + 192;

  int b = blockIdx.x >> 3;
  int h = blockIdx.x & 7;
  int wave = threadIdx.x >> 5, lane = threadIdx.x & 31;
  int hi = lane >> 4, lx = lane & 15;
  const int ch = C / HEADS;  // 24
  const bf16_t* base = qkv + (size_t)b * CO * L;
  const bf16_t* vtb  = Vt + ((size_t)b * 8 + h) * hwg * 192;

  auto qrow = [&](int cg) -> const bf16_t* {
    int f = cg / ch, cc = cg % ch;
    int fq = cross ? sim[b * 8 + f] : f;
    return base + (size_t)(qoff + h * ch + cc) * L + (size_t)fq * hwg;
  };
  auto krow = [&](int cg) -> const bf16_t* {
    int f = cg / ch, cc = cg % ch;
    return base + (size_t)(koff + h * ch + cc) * L + (size_t)f * hwg;
  };

  // Phase A: inverse L2 norms of Q and K rows
  for (int cg = wave; cg < 192; cg += 8) {
    const bf16_t* qp = qrow(cg);
    float s = 0.f;
    for (int g = lane; g < hwg; g += 32) { float v = (float)qp[g]; s += v * v; }
    for (int o = 16; o > 0; o >>= 1) s += __shfl_xor(s, o, 32);
    if (lane == 0) qn[cg] = 1.f / fmaxf(sqrtf(s), 1e-12f);
    const bf16_t* kp = krow(cg);
    float s2 = 0.f;
    for (int g = lane; g < hwg; g += 32) { float v = (float)kp[g]; s2 += v * v; }
    for (int o = 16; o > 0; o >>= 1) s2 += __shfl_xor(s2, o, 32);
    if (lane == 0) kn[cg] = 1.f / fmaxf(sqrtf(s2), 1e-12f);
  }
  __syncthreads();
  float temp = temperature[h];

  // Phase B: S(192x192) = Qg @ Kg^T, K = hwg, contiguous-K fragments
  int mt0 = (wave & 3) * 3;   // 3 M-tiles per wave
  int nt0 = (wave >> 2) * 6;  // 6 N-tiles per wave
  v8f acc[3][6];
#pragma unroll
  for (int i = 0; i < 3; ++i)
#pragma unroll
    for (int j = 0; j < 6; ++j) acc[i][j] = v8f_zero();

  const bf16_t* arow[3];
  const bf16_t* brow[6];
#pragma unroll
  for (int i = 0; i < 3; ++i) arow[i] = qrow((mt0 + i) * 16 + lx);
#pragma unroll
  for (int j = 0; j < 6; ++j) brow[j] = krow((nt0 + j) * 16 + lx);

  for (int k0 = 0; k0 < hwg; k0 += 32) {
    v16bf af[3], bfm[6];
#pragma unroll
    for (int i = 0; i < 3; ++i) af[i] = frag_contig(arow[i], k0, hi);
#pragma unroll
    for (int j = 0; j < 6; ++j) bfm[j] = frag_contig(brow[j], k0, hi);
#pragma unroll
    for (int i = 0; i < 3; ++i)
#pragma unroll
      for (int j = 0; j < 6; ++j) acc[i][j] = wmma_bf16(af[i], bfm[j], acc[i][j]);
  }
#pragma unroll
  for (int i = 0; i < 3; ++i)
#pragma unroll
    for (int j = 0; j < 6; ++j)
#pragma unroll
      for (int r = 0; r < 8; ++r) {
        int m = (mt0 + i) * 16 + hi * 8 + r;
        int n = (nt0 + j) * 16 + lx;
        S[m * 192 + n] = acc[i][j][r] * qn[m] * kn[n] * temp;
      }
  __syncthreads();

  // Phase C: softmax_1 per row: e/(sum e + exp(-max)); write P bf16 in-place
  for (int m = wave; m < 192; m += 8) {
    float* row = S + m * 192;
    float v6[6];
    float mx = -1e30f;
#pragma unroll
    for (int e = 0; e < 6; ++e) { v6[e] = row[lane + 32 * e]; mx = fmaxf(mx, v6[e]); }
    for (int o = 16; o > 0; o >>= 1) mx = fmaxf(mx, __shfl_xor(mx, o, 32));
    float sum = 0.f;
#pragma unroll
    for (int e = 0; e < 6; ++e) { v6[e] = expf(v6[e] - mx); sum += v6[e]; }
    for (int o = 16; o > 0; o >>= 1) sum += __shfl_xor(sum, o, 32);
    float inv = 1.f / (sum + expf(-mx));
    bf16_t* prow = (bf16_t*)row;
#pragma unroll
    for (int e = 0; e < 6; ++e) prow[lane + 32 * e] = (bf16_t)(v6[e] * inv);
  }
  __syncthreads();

  // Phase D: out(192 x hwg) = P(192x192) @ Vg(192 x hwg), each wave a hwg/8 strip.
  // B fragments come from Vt (g-major, contiguous over cg) -> 2x global_load_b128.
  // A fragments (P, LDS) loaded in batches of 4 so ds loads clause + single wait.
  int nspan = hwg / 8;  // 576
  for (int n0 = wave * nspan; n0 < wave * nspan + nspan; n0 += 16) {
    v8f oc[12];
#pragma unroll
    for (int mt = 0; mt < 12; ++mt) oc[mt] = v8f_zero();
    const bf16_t* vtrow = vtb + (size_t)(n0 + lx) * 192;  // this lane's column
    for (int k0 = 0; k0 < 192; k0 += 32) {
      v16bf bfm = frag_contig(vtrow, k0, hi);
#pragma unroll
      for (int mt4 = 0; mt4 < 12; mt4 += 4) {
        v16bf af4[4];
#pragma unroll
        for (int j = 0; j < 4; ++j) {
          const bf16_t* pr = (const bf16_t*)(S + ((mt4 + j) * 16 + lx) * 192);
          af4[j] = frag_contig(pr, k0, hi);  // ds_load_b128 x2, clustered
        }
#pragma unroll
        for (int j = 0; j < 4; ++j)
          oc[mt4 + j] = wmma_bf16(af4[j], bfm, oc[mt4 + j]);
      }
    }
#pragma unroll
    for (int mt = 0; mt < 12; ++mt)
#pragma unroll
      for (int r = 0; r < 8; ++r) {
        int m = mt * 16 + hi * 8 + r;
        int f = m / ch, cc = m % ch;
        int chan = h * ch + cc;
        size_t gi = ((size_t)b * C + chan) * L + (size_t)f * hwg + n0 + lx;
        float val = oc[mt][r] * para[chan];
        if (cross) y[gi] += val; else y[gi] = val;
      }
  }
}

// ---------- 9. proj GEMM + inverse-perm scatter: out[:, :, perm[l]] = Wp@y + b ----
__global__ void k_gemm_proj(const float* __restrict__ W, const float* __restrict__ bias,
                            const float* __restrict__ Yin, const int* __restrict__ vals,
                            float* __restrict__ out, int C, int L, int Lpad) {
  __shared__ bf16_t Bs[64 * 40];
  const int PITCH = 40;
  int b  = blockIdx.z;
  int n0 = blockIdx.x * 64;
  int m0 = blockIdx.y * 128;
  int wave = threadIdx.x >> 5, lane = threadIdx.x & 31;
  int hi = lane >> 4, lx = lane & 15;
  int mw = m0 + wave * 16;
  bool active = (mw < C);
  const float* Yb = Yin + (size_t)b * C * L;
  v8f acc[4];
#pragma unroll
  for (int i = 0; i < 4; ++i) acc[i] = v8f_zero();

  for (int k0 = 0; k0 < C; k0 += 32) {
    __syncthreads();
#pragma unroll
    for (int e = threadIdx.x; e < 32 * 64; e += 256) {
      int kk = e >> 6, nn = e & 63;
      Bs[nn * PITCH + kk] = (bf16_t)Yb[(size_t)(k0 + kk) * L + n0 + nn];
    }
    __syncthreads();
    if (active) {
      const float* wrow = W + (size_t)(mw + lx) * C;
      v16bf af = frag_contig_f32(wrow, k0, hi);
      v16bf bfms[4];
#pragma unroll
      for (int nt = 0; nt < 4; ++nt)
        bfms[nt] = frag_contig(Bs + (nt * 16 + lx) * PITCH, 0, hi);
#pragma unroll
      for (int nt = 0; nt < 4; ++nt)
        acc[nt] = wmma_bf16(af, bfms[nt], acc[nt]);
    }
  }
  if (active) {
#pragma unroll
    for (int nt = 0; nt < 4; ++nt) {
      int n = n0 + nt * 16 + lx;
      int p = vals[(size_t)b * Lpad + n];
#pragma unroll
      for (int r = 0; r < 8; ++r) {
        int m = mw + hi * 8 + r;
        out[((size_t)b * C + m) * L + p] = acc[nt][r] + bias[m];
      }
    }
  }
}

// ---------------------------------------------------------------------------
extern "C" void kernel_launch(void* const* d_in, const int* in_sizes, int n_in,
                              void* d_out, int out_size, void* d_ws, size_t ws_size,
                              hipStream_t stream) {
  (void)in_sizes; (void)n_in; (void)out_size; (void)ws_size;
  const int B = 2, C = 192, CO = 5 * 192, H = 192, Wd = 192;
  const int L = H * Wd;          // 36864
  const int Lpad = 65536;
  const int hwg = L / 8;         // 4608

  const float* feats  = (const float*)d_in[0];
  const float* idxm   = (const float*)d_in[1];
  const float* qkv_w  = (const float*)d_in[2];
  const float* qkv_b  = (const float*)d_in[3];
  const float* dw_w   = (const float*)d_in[4];
  const float* dw_b   = (const float*)d_in[5];
  const float* proj_w = (const float*)d_in[6];
  const float* proj_b = (const float*)d_in[7];
  const float* temp   = (const float*)d_in[8];
  const float* sa_p   = (const float*)d_in[9];
  const float* ca_p   = (const float*)d_in[10];
  float* outp = (float*)d_out;

  // workspace layout
  char* w = (char*)d_ws;
  float* keys = (float*)w;  w += (size_t)B * Lpad * 4;
  int*   vals = (int*)w;    w += (size_t)B * Lpad * 4;
  bf16_t* xb   = (bf16_t*)w; w += (size_t)B * C * L * 2;
  bf16_t* qkv0 = (bf16_t*)w; w += (size_t)B * CO * L * 2;
  bf16_t* qkv1 = (bf16_t*)w; w += (size_t)B * CO * L * 2;
  bf16_t* Vt   = (bf16_t*)w; w += (size_t)B * 8 * hwg * 192 * 2;
  float* ybuf = (float*)w;  w += (size_t)B * C * L * 4;
  float* gvm  = (float*)w;  w += (size_t)B * 8 * C * 4;
  int* sim    = (int*)w;    w += (size_t)B * 8 * 4;

  // 1-2: argsort(idx) via bitonic sort (pad to 65536 with +inf keys)
  k_init_sort<<<(B * Lpad) / 256, 256, 0, stream>>>(idxm, keys, vals, L, Lpad);
  for (int k = 2; k <= Lpad; k <<= 1)
    for (int j = k >> 1; j >= 1; j >>= 1)
      k_bitonic<<<(B * Lpad) / 256, 256, 0, stream>>>(keys, vals, j, k, Lpad);

  // 3: x = gather(feats * idx, perm) -> bf16
  k_gather<<<((size_t)B * C * L) / 256, 256, 0, stream>>>(feats, idxm, vals, xb, L, Lpad, C);

  // 4: qkv GEMM (WMMA bf16)
  dim3 gq(L / 64, (CO + 127) / 128, B);
  k_gemm_qkv<<<gq, 256, 0, stream>>>(qkv_w, qkv_b, xb, qkv0, CO, C, L);

  // 5: depthwise 3x3
  k_dwconv<<<((size_t)B * CO * L) / 256, 256, 0, stream>>>(qkv0, dw_w, dw_b, qkv1, CO, H, Wd);

  // 5b: grouped-transposed V copy for the P@V fragments
  k_vt<<<((size_t)B * 8 * hwg * 192) / 256, 256, 0, stream>>>(qkv1, Vt, C, CO, L, hwg);

  // 6-7: group means + cosine top-2 (cross-group selection)
  k_gmean<<<B * 8 * C, 128, 0, stream>>>(qkv1, gvm, C, CO, L, hwg);
  k_top2<<<B, 64, 0, stream>>>(gvm, sim, C);

  // 8: fused channel attention (in-group then cross-group accumulate)
  size_t smemB = 192 * 192 * 4 + 2 * 192 * 4;  // 148992 B
  k_attn<<<B * 8, 256, smemB, stream>>>(qkv1, Vt, temp, sa_p, sim, ybuf,
                                        C, CO, L, hwg, 0, C, 0);
  k_attn<<<B * 8, 256, smemB, stream>>>(qkv1, Vt, temp, ca_p, sim, ybuf,
                                        C, CO, L, hwg, 2 * C, 3 * C, 1);

  // 9: proj GEMM + scatter through perm into d_out
  dim3 gp(L / 64, 2, B);
  k_gemm_proj<<<gp, 256, 0, stream>>>(proj_w, proj_b, ybuf, vals, outp, C, L, Lpad);
}
